// RouteSetEncoder_13675175870829
// MI455X (gfx1250) — matrix operand order repaired and checked by reading, not compiled, not run on hardware
//
#include <hip/hip_runtime.h>
#include <hip/hip_bf16.h>

// ---------------------------------------------------------------------------
// RouteSetEncoder for MI455X (gfx1250, wave32, WMMA + TDM).
// FFN GEMMs: bf16 V_WMMA_F32_16X16X32_BF16, fp32 accumulate.
// W2 chunks staged to LDS by the Tensor Data Mover (double-buffered,
// TENSORcnt-tracked), with TDM pad_enable providing a conflict-free LDS
// stride for the B-fragment ds_load_b128 reads.
// ---------------------------------------------------------------------------

typedef __attribute__((ext_vector_type(16))) __bf16 bf16x16;
typedef __attribute__((ext_vector_type(8)))  float  floatx8;
typedef __attribute__((ext_vector_type(4)))  unsigned int uint32x4;
typedef __attribute__((ext_vector_type(8)))  int int32x8;
typedef __attribute__((ext_vector_type(4)))  int int32x4;

constexpr int kB = 64;
constexpr int kN = 512;
constexpr int kD = 512;
constexpr int kF = 2048;   // 4*D
constexpr int kR = 32;

constexpr int kTM  = 64;   // token rows per workgroup in FFN kernel
constexpr int kFC  = 64;   // F-chunk width
constexpr int kNC  = kF / kFC;   // 32 chunks
constexpr int kLDA = 520;  // LDS stride for A (bf16); 1040B = 16B multiple
constexpr int kLDH = 72;   // LDS stride for H (bf16); 144B  = 16B multiple
constexpr int kLDW = 72;   // LDS stride for W2 chunk rows (64 data + 8 pad)

// LDS layout (dynamic, offsets from base 0)
constexpr unsigned kAOff   = 0;
constexpr unsigned kABytes = (unsigned)kTM * kLDA * 2;          // 66,560
constexpr unsigned kHOff   = kAOff + kABytes;                   // 66,560
constexpr unsigned kHBytes = (unsigned)kTM * kLDH * 2;          //  9,216
constexpr unsigned kW2Off0 = kHOff + kHBytes;                   // 75,776
constexpr unsigned kW2Bytes= (unsigned)kD * kLDW * 2;           // 73,728
constexpr unsigned kW2Off1 = kW2Off0 + kW2Bytes;                // 149,504
constexpr unsigned kLdsTot = kW2Off1 + kW2Bytes;                // 223,232
constexpr unsigned kW2Elems = kW2Bytes / 2;                     // bf16 elems/buf

// --- wave (32-lane) reductions ---------------------------------------------
static __device__ __forceinline__ float wave_sum(float v) {
#pragma unroll
  for (int off = 16; off > 0; off >>= 1) v += __shfl_xor(v, off, 32);
  return v;
}
static __device__ __forceinline__ float wave_max(float v) {
#pragma unroll
  for (int off = 16; off > 0; off >>= 1) v = fmaxf(v, __shfl_xor(v, off, 32));
  return v;
}

// Load a 16-element bf16 fragment as two 16-byte vector loads.
static __device__ __forceinline__ bf16x16 ld_frag2(const __bf16* p0, const __bf16* p1) {
  union { bf16x16 v; uint4 q[2]; } u;
  u.q[0] = *(const uint4*)p0;
  u.q[1] = *(const uint4*)p1;
  return u.v;
}

static __device__ __forceinline__ floatx8 wmma_bf16(bf16x16 a, bf16x16 b, floatx8 c) {
  return __builtin_amdgcn_wmma_f32_16x16x32_bf16(false, a, false, b, (short)0, c,
                                                 false, false);
}

static __device__ __forceinline__ float gelu_exact(float x) {
  return 0.5f * x * (1.0f + erff(x * 0.70710678118654752f));
}

// ---------------------------------------------------------------------------
// TDM: DMA one W2 chunk (rows n = 0..511 of w2t, k-columns f0..f0+63) into
// LDS at ldsOff, with 16B padding after every 128B row (pad_interval=32 dw,
// pad_amount=4 dw) so LDS rows have a 144B (36-dword) stride.
// D# per CDNA5 ISA ch.8 (group0 128b, group1 256b; groups 2/3 unused for 2D).
// This toolchain's builtin takes 6 args (g0, g1, g2, g3, g4, cpol).
// ---------------------------------------------------------------------------
static __device__ __forceinline__ void tdm_load_w2_chunk(
    const __bf16* __restrict__ w2t, int f0, unsigned ldsOff) {
  unsigned long long ga = (unsigned long long)(const void*)(w2t + f0);
  uint32x4 g0;
  g0[0] = 1u;                                            // count=1, user mode
  g0[1] = ldsOff;                                        // lds_addr (bytes)
  g0[2] = (unsigned)ga;                                  // global_addr[31:0]
  g0[3] = (unsigned)((ga >> 32) & 0x1FFFFFFull)          // global_addr[56:32]
        | (2u << 30);                                    // type=2 ("image")
  int32x8 g1;
  g1[0] = (int)((1u << 16)      // data_size = 2 bytes
              | (1u << 20)      // pad_enable
              | (4u << 22)      // pad_interval: 32 dwords
              | (3u << 25));    // pad_amount: 4 dwords
  g1[1] = (int)((unsigned)(kF & 0xFFFF) << 16);          // tensor_dim0 lo16
  g1[2] = (int)(((unsigned)kF >> 16)                     // tensor_dim0 hi16
              | ((unsigned)(kD & 0xFFFF) << 16));        // tensor_dim1 lo16
  g1[3] = (int)(((unsigned)kD >> 16)                     // tensor_dim1 hi16
              | ((unsigned)kFC << 16));                  // tile_dim0 = 64
  g1[4] = kD;                                            // tile_dim1 = 512
  g1[5] = kF;                                            // dim0_stride lo32
  g1[6] = 0;
  g1[7] = 0;
  int32x4 z4 = {0, 0, 0, 0};
  int32x8 z8 = {0, 0, 0, 0, 0, 0, 0, 0};
  __builtin_amdgcn_tensor_load_to_lds(g0, g1, z4, z4, z8, 0);
}

// ---------------------------------------------------------------------------
// Kernel 1: fp32 -> bf16 weight conversion, stored transposed so WMMA
// B-fragments (KxN, lane = column) are two contiguous 16B loads.
//   w1  [D=512][F=2048]  -> w1t [F][D]   (w1t[n][k] = w1[k][n])
//   w2  [F=2048][D=512]  -> w2t [D][F]   (w2t[n][k] = w2[k][n])
// ---------------------------------------------------------------------------
__global__ __launch_bounds__(256) void convert_weights(
    const float* __restrict__ w1, const float* __restrict__ w2,
    __bf16* __restrict__ w1t, __bf16* __restrict__ w2t) {
  int idx = blockIdx.x * 256 + threadIdx.x;
  if (idx < kD * kF) {
    int k = idx / kF, n = idx % kF;             // coalesced read of w1
    w1t[(size_t)n * kD + k] = (__bf16)w1[idx];
  }
  if (idx < kF * kD) {
    int k = idx / kD, n = idx % kD;             // coalesced read of w2
    w2t[(size_t)n * kF + k] = (__bf16)w2[idx];
  }
}

// ---------------------------------------------------------------------------
// Kernel 2: fused LN -> GEMM(D,4D) -> GELU -> GEMM(4D,D) -> +residual.
// 512 threads = 16 wave32; each WG owns 64 token rows; persistent 64x512 fp32
// accumulator striped as 8 v8f C-fragments per wave. W2 chunks arrive via TDM.
// ---------------------------------------------------------------------------
__global__ __launch_bounds__(512) void ffn_kernel(
    const float* __restrict__ tokens,
    const float* __restrict__ ln1g, const float* __restrict__ ln1b,
    const __bf16* __restrict__ w1t, const float* __restrict__ b1,
    const __bf16* __restrict__ w2t, const float* __restrict__ b2,
    float* __restrict__ xout) {
  extern __shared__ __align__(16) char smem[];
  __bf16* Alds = (__bf16*)(smem + kAOff);   // [64][520] normalized tokens
  __bf16* Hlds = (__bf16*)(smem + kHOff);   // [64][72]  gelu intermediate
  __bf16* W2c0 = (__bf16*)(smem + kW2Off0); // [512][72] x2 (double buffer)

  const int tid  = threadIdx.x;
  const int wave = tid >> 5;
  const int lane = tid & 31;
  const int half = lane >> 4;   // which 16-lane half
  const int hl   = lane & 15;
  const long long rowBase = (long long)blockIdx.x * kTM;

  // Kick off TDM for chunk 0's W2 tile; overlaps with the LN prologue.
  if (wave == 0) tdm_load_w2_chunk(w2t, 0, kW2Off0);

  // ---- LayerNorm 4 rows per wave into LDS as bf16 -------------------------
#pragma unroll
  for (int i = 0; i < 4; ++i) {
    const int r = wave * 4 + i;
    const float* src = tokens + (rowBase + r) * (long long)kD;
    float v[16];
    float s = 0.f, s2 = 0.f;
#pragma unroll
    for (int j = 0; j < 16; ++j) {
      float t = src[j * 32 + lane];
      v[j] = t; s += t; s2 += t * t;
    }
    s = wave_sum(s); s2 = wave_sum(s2);
    const float mean = s * (1.0f / kD);
    const float var  = s2 * (1.0f / kD) - mean * mean;
    const float rstd = rsqrtf(var + 1e-5f);
#pragma unroll
    for (int j = 0; j < 16; ++j) {
      const int c = j * 32 + lane;
      Alds[r * kLDA + c] = (__bf16)((v[j] - mean) * rstd * ln1g[c] + ln1b[c]);
    }
  }
  __syncthreads();

  const int mt = wave & 3;    // row tile (both stages)
  const int nt = wave >> 2;   // stage-1 column tile within the F-chunk
  const int cg = wave >> 2;   // stage-2 column group (128 cols)

  floatx8 xacc[8];
#pragma unroll
  for (int t = 0; t < 8; ++t) xacc[t] = (floatx8){0.f,0.f,0.f,0.f,0.f,0.f,0.f,0.f};

  // A-fragment row pointer: lane holds row M = lane%16, K split by lane half.
  const __bf16* arow = Alds + (mt * 16 + hl) * kLDA + half * 8;

  for (int chunk = 0; chunk < kNC; ++chunk) {
    const int f0 = chunk * kFC;

    // Prefetch next chunk's W2 tile into the other buffer (wave 0 only).
    if (wave == 0 && (chunk + 1) < kNC)
      tdm_load_w2_chunk(w2t, f0 + kFC,
                        ((chunk + 1) & 1) ? kW2Off1 : kW2Off0);

    // ---- stage 1: H = gelu(A @ W1[:, f0:f0+64] + b1), one 16x16 tile/wave --
    floatx8 hacc = (floatx8){0.f,0.f,0.f,0.f,0.f,0.f,0.f,0.f};
    {
      const __bf16* bptr = w1t + (size_t)(f0 + nt * 16 + hl) * kD + half * 8;
#pragma unroll 4
      for (int k0 = 0; k0 < kD; k0 += 32) {
        bf16x16 a = ld_frag2(arow + k0, arow + k0 + 16);
        bf16x16 b = ld_frag2(bptr + k0, bptr + k0 + 16);
        hacc = wmma_bf16(a, b, hacc);
      }
    }
    const float b1v = b1[f0 + nt * 16 + hl];
#pragma unroll
    for (int vv = 0; vv < 8; ++vv) {
      // C layout: VGPR v -> row v + 8*half, col = lane%16
      float g = gelu_exact(hacc[vv] + b1v);
      Hlds[(mt * 16 + vv + half * 8) * kLDH + nt * 16 + hl] = (__bf16)g;
    }

    // Ensure chunk's W2 TDM transfer has landed (TDM ops complete in order:
    // with the next chunk in flight, waiting for <=1 retires this chunk's).
    if (wave == 0) {
      if ((chunk + 1) < kNC) __builtin_amdgcn_s_wait_tensorcnt(1);
      else                   __builtin_amdgcn_s_wait_tensorcnt(0);
    }
    __syncthreads();  // H chunk + W2 chunk visible to all waves

    // ---- stage 2: Xacc += H @ W2[f0:f0+64, :]  (B frags from LDS) ---------
    {
      const __bf16* ha = Hlds + (mt * 16 + hl) * kLDH + half * 8;
      bf16x16 a0 = ld_frag2(ha,      ha + 16);  // K = 0..31
      bf16x16 a1 = ld_frag2(ha + 32, ha + 48);  // K = 32..63
      const __bf16* wc = W2c0 + (size_t)(chunk & 1) * kW2Elems;
#pragma unroll
      for (int ct = 0; ct < 8; ++ct) {
        const __bf16* bp = wc + (cg * 128 + ct * 16 + hl) * kLDW + half * 8;
        bf16x16 b0  = ld_frag2(bp,      bp + 16);
        bf16x16 b1f = ld_frag2(bp + 32, bp + 48);
        xacc[ct] = wmma_bf16(a0, b0, xacc[ct]);
        xacc[ct] = wmma_bf16(a1, b1f, xacc[ct]);
      }
    }
    __syncthreads();  // stage-2 done; next chunk may overwrite H / W2 buffer
  }

  // ---- epilogue: + b2 + residual, store fp32 x ----------------------------
#pragma unroll
  for (int ct = 0; ct < 8; ++ct) {
    const int col = cg * 128 + ct * 16 + hl;
    const float b2v = b2[col];
#pragma unroll
    for (int vv = 0; vv < 8; ++vv) {
      const long long row = rowBase + mt * 16 + vv + half * 8;
      xout[row * kD + col] = xacc[ct][vv] + b2v + tokens[row * kD + col];
    }
  }
}

// ---------------------------------------------------------------------------
// Kernel 3: tok_logits[b,n] = LN(x[b,n,:]) . wp_w + wp_bias  (one wave/row)
// ---------------------------------------------------------------------------
__global__ __launch_bounds__(256) void logits_kernel(
    const float* __restrict__ x, const float* __restrict__ g,
    const float* __restrict__ bb, const float* __restrict__ w,
    const float* __restrict__ bias, float* __restrict__ out) {
  const int wave = threadIdx.x >> 5, lane = threadIdx.x & 31;
  const long long row = (long long)blockIdx.x * 8 + wave;
  const float* src = x + row * kD;
  float v[16];
  float s = 0.f, s2 = 0.f;
#pragma unroll
  for (int j = 0; j < 16; ++j) {
    float t = src[j * 32 + lane];
    v[j] = t; s += t; s2 += t * t;
  }
  s = wave_sum(s); s2 = wave_sum(s2);
  const float mean = s * (1.0f / kD);
  const float rstd = rsqrtf(s2 * (1.0f / kD) - mean * mean + 1e-5f);
  float acc = 0.f;
#pragma unroll
  for (int j = 0; j < 16; ++j) {
    const int c = j * 32 + lane;
    acc += ((v[j] - mean) * rstd * g[c] + bb[c]) * w[c];
  }
  acc = wave_sum(acc);
  if (lane == 0) out[row] = acc + bias[0];
}

// ---------------------------------------------------------------------------
// Kernel 4: within-route masked softmax pool. One WG per (b, route).
// ---------------------------------------------------------------------------
__global__ __launch_bounds__(256) void route_pool_kernel(
    const float* __restrict__ x, const float* __restrict__ logits,
    const unsigned char* __restrict__ token_mask,
    const int* __restrict__ route_ids,
    float* __restrict__ routes, float* __restrict__ present) {
  const int b = blockIdx.x >> 5;
  const int r = blockIdx.x & 31;
  const int tid = threadIdx.x, wave = tid >> 5, lane = tid & 31;
  __shared__ float redA[8];
  __shared__ float redB[8];
  __shared__ float wgt[kN];
  __shared__ float s_max, s_inv;
  __shared__ int   s_cnt;

  const float* lrow = logits + (long long)b * kN;

  // pass 1: masked max + member count
  float mx = -1e30f, cntf = 0.f;
  for (int n = tid; n < kN; n += 256) {
    const bool mem = (route_ids[n] == r) && (token_mask[b * kN + n] != 0);
    if (mem) { mx = fmaxf(mx, lrow[n]); cntf += 1.f; }
  }
  mx = wave_max(mx); cntf = wave_sum(cntf);
  if (lane == 0) { redA[wave] = mx; redB[wave] = cntf; }
  __syncthreads();
  if (tid == 0) {
    float m = -1e30f, c = 0.f;
    for (int i = 0; i < 8; ++i) { m = fmaxf(m, redA[i]); c += redB[i]; }
    s_max = m; s_cnt = (int)c;
  }
  __syncthreads();

  // pass 2: exp weights + sum
  const float m = s_max;
  float se = 0.f;
  for (int n = tid; n < kN; n += 256) {
    const bool mem = (route_ids[n] == r) && (token_mask[b * kN + n] != 0);
    const float e = mem ? expf(lrow[n] - m) : 0.f;
    wgt[n] = e;
    se += e;
  }
  se = wave_sum(se);
  if (lane == 0) redA[wave] = se;
  __syncthreads();
  if (tid == 0) {
    float s = 0.f;
    for (int i = 0; i < 8; ++i) s += redA[i];
    s_inv = (s_cnt > 0) ? 1.f / fmaxf(s, 1e-6f) : 0.f;
  }
  __syncthreads();

  // pass 3: weighted row accumulation, 2 columns per thread
  const float inv = s_inv;
  float a0 = 0.f, a1 = 0.f;
  const float* xb = x + (long long)b * kN * kD;
  for (int n = 0; n < kN; ++n) {
    const float w = wgt[n];             // uniform across the block
    if (w != 0.f) {
      const float* xr = xb + (long long)n * kD;
      a0 += w * xr[tid];
      a1 += w * xr[tid + 256];
    }
  }
  const long long ro = ((long long)b * kR + r) * kD;
  routes[ro + tid]       = a0 * inv;
  routes[ro + tid + 256] = a1 * inv;
  if (tid == 0) present[b * kR + r] = (s_cnt > 0) ? 1.f : 0.f;
}

// ---------------------------------------------------------------------------
// Kernel 5: route-level LN + softmax pool -> out[b, :D]. One WG per batch.
// ---------------------------------------------------------------------------
__global__ __launch_bounds__(256) void final_kernel(
    const float* __restrict__ routes, const float* __restrict__ present,
    const float* __restrict__ g, const float* __restrict__ bb,
    const float* __restrict__ w, const float* __restrict__ bias,
    float* __restrict__ out) {
  const int b = blockIdx.x;
  const int tid = threadIdx.x, wave = tid >> 5, lane = tid & 31;
  __shared__ float rl[kR];
  __shared__ float rw[kR];

#pragma unroll
  for (int i = 0; i < 4; ++i) {
    const int r = wave * 4 + i;
    const float* src = routes + ((long long)b * kR + r) * kD;
    float v[16];
    float s = 0.f, s2 = 0.f;
#pragma unroll
    for (int j = 0; j < 16; ++j) {
      float t = src[j * 32 + lane];
      v[j] = t; s += t; s2 += t * t;
    }
    s = wave_sum(s); s2 = wave_sum(s2);
    const float mean = s * (1.0f / kD);
    const float rstd = rsqrtf(s2 * (1.0f / kD) - mean * mean + 1e-5f);
    float acc = 0.f;
#pragma unroll
    for (int j = 0; j < 16; ++j) {
      const int c = j * 32 + lane;
      acc += ((v[j] - mean) * rstd * g[c] + bb[c]) * w[c];
    }
    acc = wave_sum(acc);
    if (lane == 0)
      rl[r] = (present[b * kR + r] > 0.f) ? (acc + bias[0]) : -1e30f;
  }
  __syncthreads();

  if (tid < kR) {  // one wave does the 32-wide softmax
    const float l = rl[tid];
    float mmax = wave_max(l);
    const float e = expf(l - mmax);
    float ssum = wave_sum(e);
    float wv = (present[b * kR + tid] > 0.f) ? e / fmaxf(ssum, 1e-6f) : 0.f;
    rw[tid] = wv;
  }
  __syncthreads();

  float a0 = 0.f, a1 = 0.f;
#pragma unroll
  for (int r = 0; r < kR; ++r) {
    const float wv = rw[r];
    const float* rr = routes + ((long long)b * kR + r) * kD;
    a0 += wv * rr[tid];
    a1 += wv * rr[tid + 256];
  }
  out[(long long)b * kD + tid]       = a0;
  out[(long long)b * kD + tid + 256] = a1;
}

// ---------------------------------------------------------------------------
extern "C" void kernel_launch(void* const* d_in, const int* in_sizes, int n_in,
                              void* d_out, int out_size, void* d_ws, size_t ws_size,
                              hipStream_t stream) {
  const float*         tokens     = (const float*)d_in[0];
  const unsigned char* token_mask = (const unsigned char*)d_in[1];
  const int*           route_ids  = (const int*)d_in[2];
  const float* ln1g   = (const float*)d_in[3];
  const float* ln1b   = (const float*)d_in[4];
  const float* w1     = (const float*)d_in[5];
  const float* b1     = (const float*)d_in[6];
  const float* w2     = (const float*)d_in[7];
  const float* b2     = (const float*)d_in[8];
  const float* wpg    = (const float*)d_in[9];
  const float* wpb    = (const float*)d_in[10];
  const float* wpw    = (const float*)d_in[11];
  const float* wpbias = (const float*)d_in[12];
  const float* rpg    = (const float*)d_in[13];
  const float* rpb    = (const float*)d_in[14];
  const float* rpw    = (const float*)d_in[15];
  const float* rpbias = (const float*)d_in[16];

  char* ws = (char*)d_ws;
  __bf16* w1t    = (__bf16*)ws; ws += (size_t)kF * kD * sizeof(__bf16);
  __bf16* w2t    = (__bf16*)ws; ws += (size_t)kD * kF * sizeof(__bf16);
  float*  x      = (float*)ws;  ws += (size_t)kB * kN * kD * sizeof(float);
  float*  logits = (float*)ws;  ws += (size_t)kB * kN * sizeof(float);
  float*  routes = (float*)ws;  ws += (size_t)kB * kR * kD * sizeof(float);
  float*  present= (float*)ws;  ws += (size_t)kB * kR * sizeof(float);

  // 1) one-time bf16 transposed weights
  convert_weights<<<(kD * kF + 255) / 256, 256, 0, stream>>>(w1, w2, w1t, w2t);

  // 2) fused FFN (WMMA + TDM): 512 WGs of 64 token rows, 16 waves each
  ffn_kernel<<<(kB * kN) / kTM, 512, kLdsTot, stream>>>(
      tokens, ln1g, ln1b, w1t, b1, w2t, b2, x);

  // 3) token logits
  logits_kernel<<<(kB * kN) / 8, 256, 0, stream>>>(x, wpg, wpb, wpw, wpbias, logits);

  // 4) within-route masked softmax pool
  route_pool_kernel<<<kB * kR, 256, 0, stream>>>(x, logits, token_mask, route_ids,
                                                 routes, present);

  // 5) route-level pool -> output [64, 512]
  final_kernel<<<kB, 256, 0, stream>>>(routes, present, rpg, rpb, rpw, rpbias,
                                       (float*)d_out);
}
